// SelfAttention_77738908058069
// MI455X (gfx1250) — matrix operand order repaired and checked by reading, not compiled, not run on hardware
//
#include <hip/hip_runtime.h>

typedef __attribute__((ext_vector_type(16))) _Float16 v16h;
typedef __attribute__((ext_vector_type(8)))  float    v8f;

#define BM 128
#define BN 128
#define BK 32
#define APITCH 40   // f16 elems per LDS row (80B = 20 dwords) -> conflict-friendly
#define BPITCH 40

union V16U { v16h v; uint4 q[2]; };

// raw global-load of 4 source elements (fp32: 16B, f16: 8B)
__device__ __forceinline__ float4 ldg4(const float* p)    { return *(const float4*)p; }
__device__ __forceinline__ uint2  ldg4(const _Float16* p) { return *(const uint2*)p; }
// convert/copy 4 elements to f16 and store (8B ds_store_b64)
__device__ __forceinline__ void cvtst4(const float4& v, _Float16* d) {
  d[0] = (_Float16)v.x; d[1] = (_Float16)v.y;
  d[2] = (_Float16)v.z; d[3] = (_Float16)v.w;
}
__device__ __forceinline__ void cvtst4(const uint2& v, _Float16* d) { *(uint2*)d = v; }

__device__ __forceinline__ void stout(float v, float* p)    { *p = v; }
__device__ __forceinline__ void stout(float v, _Float16* p) { *p = (_Float16)v; }

// C = scale * (A[M,K] * Bsrc^T + bias), with Bsrc f16 [N,K] (always NT/fast path).
//   TRANSC: store C transposed: C[n*ldc + m] (used to produce V^T).
//   CSKIP : skip blocks fully above causal diagonal (scores).
//   CKLIM : clip K loop at the diagonal (P*V).
// Block: 256 thr = 8 waves, tile 128x128; wave tile 32x64 = 8 WMMA accums.
// Double-buffered LDS, one barrier per k-step; global loads for tile k+1
// are issued before the WMMAs of tile k.
template<typename AT, typename OT, bool TRANSC, bool CSKIP, bool CKLIM>
__global__ __launch_bounds__(256) void gemm_wmma(
    const AT* __restrict__ A, int lda, long batchA,
    const _Float16* __restrict__ B, int ldb, long batchB,
    OT* __restrict__ C, int ldc, long batchC,
    const float* __restrict__ bias, float scale, int K)
{
  const int m0 = blockIdx.y * BM;
  const int n0 = blockIdx.x * BN;
  if (CSKIP && n0 > m0 + (BM - 1)) return;   // fully masked causal tile

  A += (size_t)blockIdx.z * batchA;
  B += (size_t)blockIdx.z * batchB;
  C += (size_t)blockIdx.z * batchC;

  __shared__ _Float16 sA[2][BM * APITCH];   // [row m][k]
  __shared__ _Float16 sB[2][BN * BPITCH];   // [col n][k]

  const int tid  = threadIdx.x;
  const int lane = tid & 31;
  const int wave = tid >> 5;
  const int wr = wave & 3;     // wave M index (4 waves * 32 rows)
  const int wc = wave >> 2;    // wave N index (2 waves * 64 cols)
  const int fl = lane & 15;
  const int fh = lane >> 4;

  int Keff = K;
  if (CKLIM) { int lim = m0 + BM; Keff = (K < lim) ? K : lim; }
  const int nk = (Keff + BK - 1) / BK;

  using ARaw = decltype(ldg4((const AT*)nullptr));

  auto loadA = [&](int kt, ARaw* ar, bool pf) {
    #pragma unroll
    for (int t = 0; t < 4; ++t) {               // 128x32 = 1024 groups of 4
      int g = tid + t * 256;
      int row = g >> 3, c4 = (g & 7) * 4;
      const AT* p = A + (size_t)(m0 + row) * lda + kt * BK + c4;
      ar[t] = ldg4(p);
      if (pf) __builtin_prefetch(p + BK, 0, 3);
    }
  };
  auto storeA = [&](const ARaw* ar, _Float16* dst) {
    #pragma unroll
    for (int t = 0; t < 4; ++t) {
      int g = tid + t * 256;
      int row = g >> 3, c4 = (g & 7) * 4;
      cvtst4(ar[t], &dst[row * APITCH + c4]);
    }
  };
  auto loadB = [&](int kt, uint2* br, bool pf) {
    #pragma unroll
    for (int t = 0; t < 4; ++t) {               // 128x32 = 1024 groups of 4
      int g = tid + t * 256;
      int n = g >> 3, c4 = (g & 7) * 4;
      const _Float16* p = B + (size_t)(n0 + n) * ldb + kt * BK + c4;
      br[t] = ldg4(p);
      if (pf) __builtin_prefetch(p + BK, 0, 3);
    }
  };
  auto storeB = [&](const uint2* br, _Float16* dst) {
    #pragma unroll
    for (int t = 0; t < 4; ++t) {
      int g = tid + t * 256;
      int n = g >> 3, c4 = (g & 7) * 4;
      cvtst4(br[t], &dst[n * BPITCH + c4]);
    }
  };

  v8f acc[2][4];
  #pragma unroll
  for (int mf = 0; mf < 2; ++mf)
    #pragma unroll
    for (int f = 0; f < 4; ++f)
      #pragma unroll
      for (int r = 0; r < 8; ++r) acc[mf][f][r] = 0.0f;

  ARaw ar[4]; uint2 br[4];
  loadA(0, ar, nk > 1);
  loadB(0, br, nk > 1);

  for (int kt = 0; kt < nk; ++kt) {
    _Float16* curA = sA[kt & 1];
    _Float16* curB = sB[kt & 1];
    storeA(ar, curA);
    storeB(br, curB);
    __syncthreads();
    if (kt + 1 < nk) {                     // overlap next-tile loads with WMMA
      loadA(kt + 1, ar, kt + 2 < nk);
      loadB(kt + 1, br, kt + 2 < nk);
    }

    // A fragments (16x32 f16): lanes 0-15 hold k=[0..7],[16..23]; 16-31 +8
    V16U a[2];
    #pragma unroll
    for (int mf = 0; mf < 2; ++mf) {
      const _Float16* pa = &curA[(wr * 32 + mf * 16 + fl) * APITCH + 8 * fh];
      a[mf].q[0] = *(const uint4*)pa;          // k = 8h .. 8h+7
      a[mf].q[1] = *(const uint4*)(pa + 16);   // k = 8h+16 .. 8h+23
    }
    // B fragments (32x16 f16) + 8 WMMAs
    #pragma unroll
    for (int f = 0; f < 4; ++f) {
      V16U bf;
      const _Float16* pb = &curB[(wc * 64 + f * 16 + fl) * BPITCH + 8 * fh];
      bf.q[0] = *(const uint4*)pb;             // k = 8h .. 8h+7
      bf.q[1] = *(const uint4*)(pb + 16);      // k = 8h+16 .. 8h+23
      acc[0][f] = __builtin_amdgcn_wmma_f32_16x16x32_f16(
          false, a[0].v, false, bf.v, (short)0, acc[0][f], false, false);
      acc[1][f] = __builtin_amdgcn_wmma_f32_16x16x32_f16(
          false, a[1].v, false, bf.v, (short)0, acc[1][f], false, false);
    }
    // next iteration writes the other LDS buffer, last read two iterations
    // ago -- the single barrier above is sufficient.
  }

  // ---- epilogue: C layout lane l: N = l%16, M = 8*(l/16)+r ----
  #pragma unroll
  for (int mf = 0; mf < 2; ++mf) {
    #pragma unroll
    for (int f = 0; f < 4; ++f) {
      const int gn = n0 + wc * 64 + f * 16 + fl;
      const float bv = bias ? bias[gn] : 0.0f;
      #pragma unroll
      for (int r = 0; r < 8; ++r) {
        const int gm = m0 + wr * 32 + mf * 16 + fh * 8 + r;
        const float v = (acc[mf][f][r] + bv) * scale;
        if (TRANSC) stout(v, &C[(size_t)gn * ldc + gm]);
        else        stout(v, &C[(size_t)gm * ldc + gn]);
      }
    }
  }
}

// Transpose + fp32->f16 convert: Wt[n*K + k] = (f16)W[k*N + n].
// 64x64 tiles through LDS, both sides coalesced.
__global__ __launch_bounds__(256) void transcvt(
    const float* __restrict__ W, _Float16* __restrict__ Wt, int K, int N) {
  __shared__ float tile[64][65];
  const int tid = threadIdx.x;
  const int k0 = blockIdx.y * 64, n0 = blockIdx.x * 64;
  #pragma unroll
  for (int t = 0; t < 16; ++t) {
    int idx = tid + t * 256;
    int r = idx >> 6, c = idx & 63;          // r: k-local, c: n-local
    tile[r][c] = W[(size_t)(k0 + r) * N + n0 + c];
  }
  __syncthreads();
  #pragma unroll
  for (int t = 0; t < 16; ++t) {
    int idx = tid + t * 256;
    int r = idx >> 6, c = idx & 63;          // r: n-local, c: k-local
    Wt[(size_t)(n0 + r) * K + k0 + c] = (_Float16)tile[c][r];
  }
}

// Row-wise causal softmax over S[B*T, T] (fp32), writing P (f16) in place.
__global__ __launch_bounds__(256) void softmax_causal(float* __restrict__ S, int T) {
  const int row = blockIdx.x;       // b*T + i
  const int i   = row & (T - 1);    // T is a power of two
  float*     srow = S + (size_t)row * T;
  _Float16*  prow = (_Float16*)srow;   // in-place: all reads precede barriers
  const int tid = threadIdx.x;
  __shared__ float red[256];

  float vals[8];
  float m = -3.0e38f;
  #pragma unroll
  for (int e = 0; e < 8; ++e) {
    int j = e * 256 + tid;
    float x = (j <= i) ? srow[j] : -3.0e38f;
    vals[e] = x;
    m = fmaxf(m, x);
  }
  red[tid] = m; __syncthreads();
  for (int s = 128; s > 0; s >>= 1) {
    if (tid < s) red[tid] = fmaxf(red[tid], red[tid + s]);
    __syncthreads();
  }
  const float rmax = red[0];
  __syncthreads();

  float sum = 0.0f;
  #pragma unroll
  for (int e = 0; e < 8; ++e) {
    int j = e * 256 + tid;
    float ex = (j <= i) ? __expf(vals[e] - rmax) : 0.0f;
    vals[e] = ex; sum += ex;
  }
  red[tid] = sum; __syncthreads();
  for (int s = 128; s > 0; s >>= 1) {
    if (tid < s) red[tid] += red[tid + s];
    __syncthreads();
  }
  const float inv = 1.0f / red[0];

  #pragma unroll
  for (int e = 0; e < 8; ++e) {
    int j = e * 256 + tid;
    prow[j] = (_Float16)(vals[e] * inv);
  }
}

extern "C" void kernel_launch(void* const* d_in, const int* in_sizes, int n_in,
                              void* d_out, int out_size, void* d_ws, size_t ws_size,
                              hipStream_t stream) {
  (void)in_sizes; (void)n_in; (void)out_size; (void)ws_size;
  const float* y  = (const float*)d_in[0];
  const float* Wq = (const float*)d_in[1];
  const float* bq = (const float*)d_in[2];
  const float* Wk = (const float*)d_in[3];
  const float* bk = (const float*)d_in[4];
  const float* Wv = (const float*)d_in[5];
  const float* bv = (const float*)d_in[6];
  const float* Wo = (const float*)d_in[7];
  const float* bo = (const float*)d_in[8];
  float* out = (float*)d_out;

  const int Bb = 4, T = 2048, H = 1024, DK = 1024, DV = 1024;
  const int MT = Bb * T;   // 8192 rows with batch folded

  // workspace layout (136 MB total)
  _Float16* Qh  = (_Float16*)d_ws;                      // 16 MB  [MT,DK]
  _Float16* Kh  = Qh + (size_t)MT * DK;                 // 16 MB  [MT,DK]
  _Float16* Vt  = Kh + (size_t)MT * DK;                 // 16 MB  [B,DV,T] (V^T)
  float*    S   = (float*)(Vt + (size_t)MT * DV);       // 64 MB  (reused as P f16)
  _Float16* Ch  = (_Float16*)(S + (size_t)Bb * T * T);  // 16 MB  [MT,DV]
  _Float16* Wqt = Ch + (size_t)MT * DV;                 // 2 MB   [DK,H]
  _Float16* Wkt = Wqt + (size_t)DK * H;                 // 2 MB
  _Float16* Wvt = Wkt + (size_t)DK * H;                 // 2 MB
  _Float16* Wot = Wvt + (size_t)DV * H;                 // 2 MB   [H,DV]

  dim3 blk(256);

  // 0) one-time weight transpose + f16 convert (all GEMMs take the NT fast path)
  dim3 gt(16, 16);
  transcvt<<<gt, blk, 0, stream>>>(Wq, Wqt, H, DK);
  transcvt<<<gt, blk, 0, stream>>>(Wk, Wkt, H, DK);
  transcvt<<<gt, blk, 0, stream>>>(Wv, Wvt, H, DV);
  transcvt<<<gt, blk, 0, stream>>>(Wo, Wot, DV, H);

  // 1) Q/K projections (fp32 -> f16), Q pre-scaled by 1/sqrt(DK)=1/32
  dim3 gproj(DK / BN, MT / BM, 1);
  gemm_wmma<float, _Float16, false, false, false><<<gproj, blk, 0, stream>>>(
      y, H, 0L, Wqt, H, 0L, Qh, DK, 0L, bq, 0.03125f, H);
  gemm_wmma<float, _Float16, false, false, false><<<gproj, blk, 0, stream>>>(
      y, H, 0L, Wkt, H, 0L, Kh, DK, 0L, bk, 1.0f, H);
  //    V projection written transposed: Vt[b][d][t]
  gemm_wmma<float, _Float16, true, false, false><<<dim3(DV / BN, T / BM, Bb), blk, 0, stream>>>(
      y, H, (long)T * H, Wvt, H, 0L, Vt, T, (long)DV * T, bv, 1.0f, H);

  // 2) S = Q * K^T  (NT GEMM, causal tiles skipped), fp32 out
  gemm_wmma<_Float16, float, false, true, false><<<dim3(T / BN, T / BM, Bb), blk, 0, stream>>>(
      Qh, DK, (long)T * DK, Kh, DK, (long)T * DK, S, T, (long)T * T, nullptr, 1.0f, DK);

  // 3) causal softmax, P (f16) written in place over S
  softmax_causal<<<dim3(Bb * T), blk, 0, stream>>>(S, T);

  // 4) ctx = P * V = P * (V^T)^T (NT via Vt), K loop clipped at diagonal, f16 out
  gemm_wmma<_Float16, _Float16, false, false, true><<<dim3(DV / BN, T / BM, Bb), blk, 0, stream>>>(
      (const _Float16*)S, 2 * T, (long)T * 2 * T, Vt, T, (long)DV * T,
      Ch, DV, (long)T * DV, nullptr, 1.0f, T);

  // 5) out = ctx * Wo + bo (NT via Wot), fp32 out
  gemm_wmma<_Float16, float, false, false, false><<<dim3(H / BN, MT / BM, 1), blk, 0, stream>>>(
      Ch, DV, 0L, Wot, DV, 0L, out, H, 0L, bo, 1.0f, DV);
}
